// GraphTransformerAttn_80238579023946
// MI455X (gfx1250) — matrix operand order repaired and checked by reading, not compiled, not run on hardware
//
#include <hip/hip_runtime.h>
#include <hip/hip_bf16.h>
#include <math.h>

// ---------------------------------------------------------------------------
// Graph transformer attention for MI455X (gfx1250, wave32).
//   N=50000 nodes, E=800000 edges, DIM=256, HEADS=8, DIM_OUT=32 (inner=256)
// Phase 0: one-time bf16 repack (x -> bf16 row-major; W -> bf16 [n][k] with
//          the 1/sqrt(32) score scale folded into Wq/bq).
// Phase 1: Q/K/V projections via v_wmma_f32_16x16x32_bf16; inner loop is
//          4x global_load_b128 + 1 WMMA per K=32 step.
// Phase 2: edge scores + segment softmax + weighted scatter (L2-resident
//          gathers, float atomics).
// ---------------------------------------------------------------------------

#define N_NODES 50000
#define N_EDGES 800000
#define DIM     256
#define HEADS   8
#define DIM_OUT 32
#define INNER   (HEADS * DIM_OUT)   // 256

typedef __attribute__((ext_vector_type(16))) __bf16 v16bf;
typedef __attribute__((ext_vector_type(8)))  __bf16 v8bf;
typedef __attribute__((ext_vector_type(4)))  __bf16 v4bf;
typedef __attribute__((ext_vector_type(8)))  float  v8f;

// ---------------------------------------------------------------------------
// Init: out = 0, z = 0, m = -inf
// ---------------------------------------------------------------------------
__global__ void gta_init_kernel(float* __restrict__ out,
                                float* __restrict__ mmax,
                                float* __restrict__ zsum) {
  int t = blockIdx.x * blockDim.x + threadIdx.x;
  if (t < N_NODES * INNER) out[t] = 0.0f;
  if (t < N_NODES * HEADS) {
    mmax[t] = -INFINITY;
    zsum[t] = 0.0f;
  }
}

// ---------------------------------------------------------------------------
// x (f32 row-major) -> xb (bf16 row-major).  4 elements per thread.
// ---------------------------------------------------------------------------
__global__ void __launch_bounds__(256)
gta_cvt_x_kernel(const float* __restrict__ x, __bf16* __restrict__ xb) {
  int t = blockIdx.x * blockDim.x + threadIdx.x;      // quad index
  if (t >= (N_NODES * DIM) / 4) return;
  const float4 f = ((const float4*)x)[t];
  v4bf o;
  o[0] = (__bf16)f.x; o[1] = (__bf16)f.y; o[2] = (__bf16)f.z; o[3] = (__bf16)f.w;
  ((v4bf*)xb)[t] = o;
}

// ---------------------------------------------------------------------------
// W[k][n] (f32) -> Wt[mat][n][k] (bf16), with 1/sqrt(DIM_OUT) folded into Q.
// Also emits scaled biases.
// ---------------------------------------------------------------------------
__global__ void __launch_bounds__(256)
gta_prep_w_kernel(const float* __restrict__ Wq, const float* __restrict__ bq,
                  const float* __restrict__ Wk, const float* __restrict__ bk,
                  const float* __restrict__ Wv, const float* __restrict__ bv,
                  __bf16* __restrict__ Wt, float* __restrict__ bsc) {
  int t = blockIdx.x * blockDim.x + threadIdx.x;      // mat*65536 + n*256 + k
  if (t >= 3 * DIM * INNER) return;
  const int mat = t >> 16;
  const int r   = t & 0xFFFF;
  const int n   = r >> 8;
  const int kk  = r & 0xFF;
  const float* W = (mat == 0) ? Wq : (mat == 1) ? Wk : Wv;
  const float scale = (mat == 0) ? 0.17677669529663687f : 1.0f;
  Wt[t] = (__bf16)(W[(size_t)kk * INNER + n] * scale);
  if (kk == 0) {
    const float* b = (mat == 0) ? bq : (mat == 1) ? bk : bv;
    bsc[mat * INNER + n] = b[n] * scale;
  }
}

// ---------------------------------------------------------------------------
// Fused QKV GEMM: one wave computes one 16x16 tile of (xb @ Wt^T + b).
//   grid = (M/16, 4, 3), block = 128 (4 waves; wave w -> ntile = y*4+w).
// A fragment (16x32 bf16): lane {g=l>>4, m=l&15}: element idx
//   0..7  -> K = k0 + 16g    + idx      (16 contiguous bytes)
//   8..15 -> K = k0 + 16g + idx   (16 contiguous bytes)
// B fragment (32x16 bf16): lane {g, n}: element idx -> K = k0 + 16g + idx
//   (32 contiguous bytes in Wt's k-major row for column col0+n).
// C/D (16x16 f32): lane {g, n}, VGPR v -> element (M = v + 8g, N = n).
// ---------------------------------------------------------------------------
__global__ void __launch_bounds__(128)
gta_qkv_gemm_kernel(const __bf16* __restrict__ xb,
                    const __bf16* __restrict__ Wt,
                    const float* __restrict__ bsc,
                    float* __restrict__ q, float* __restrict__ k,
                    float* __restrict__ v) {
  const int wave = threadIdx.x >> 5;
  const int lane = threadIdx.x & 31;
  const int g    = lane >> 4;       // half-wave
  const int m    = lane & 15;       // A row / B,C column within tile

  const int mtile = blockIdx.x;
  const int ntile = blockIdx.y * 4 + wave;
  const int mat   = blockIdx.z;     // 0=Q 1=K 2=V

  const int row0 = mtile * 16;
  const int col0 = ntile * 16;

  const __bf16* arow = xb + (size_t)(row0 + m) * DIM;
  const __bf16* brow = Wt + (size_t)mat * DIM * INNER + (size_t)(col0 + m) * DIM;
  float*        outp = (mat == 0) ? q : (mat == 1) ? k : v;

  v8f acc = {};
#pragma unroll
  for (int kk = 0; kk < DIM / 32; ++kk) {
    const int k0 = kk * 32;
    const v8bf a_lo = *(const v8bf*)(arow + k0 + 16 * g);       // K = k0+16g ..+7
    const v8bf a_hi = *(const v8bf*)(arow + k0 + 16 * g + 8);   // K = k0+16g+8 ..+15
    const v16bf A = __builtin_shufflevector(a_lo, a_hi,
        0, 1, 2, 3, 4, 5, 6, 7, 8, 9, 10, 11, 12, 13, 14, 15);
    const v16bf B = *(const v16bf*)(brow + k0 + 16 * g);        // K = k0+16g ..+15
    acc = __builtin_amdgcn_wmma_f32_16x16x32_bf16(
        /*neg_a=*/false, A, /*neg_b=*/false, B,
        /*c_mod=*/(short)0, acc, /*reuse_a=*/false, /*reuse_b=*/false);
  }

  const float bias = bsc[mat * INNER + col0 + m];
#pragma unroll
  for (int vv = 0; vv < 8; ++vv) {
    const int row = row0 + vv + 8 * g;
    outp[(size_t)row * INNER + col0 + m] = acc[vv] + bias;
  }
}

// ---------------------------------------------------------------------------
// Sign-aware float atomic max (mixed signs incl. -0.0).
// ---------------------------------------------------------------------------
__device__ __forceinline__ void atomicMaxF32(float* addr, float val) {
  if ((__float_as_uint(val) >> 31) == 0u) {
    atomicMax((int*)addr, __float_as_int(val));           // non-negative
  } else {
    atomicMin((unsigned int*)addr, __float_as_uint(val)); // negative
  }
}

// ---------------------------------------------------------------------------
// Edge scores: wave per edge.  Lane l covers dims 8l..8l+7 (head h = l/4).
// score[e][h] = dot(q[src], k[dst]) per head (q pre-scaled by 1/sqrt(32)).
// Atomic segment-max into m[dst][h].
// ---------------------------------------------------------------------------
__global__ void __launch_bounds__(256)
gta_edge_scores_kernel(const float* __restrict__ q, const float* __restrict__ k,
                       const int* __restrict__ src, const int* __restrict__ dst,
                       float* __restrict__ scores, float* __restrict__ mmax) {
  const int e = blockIdx.x * (blockDim.x >> 5) + (threadIdx.x >> 5);
  if (e >= N_EDGES) return;
  const int lane = threadIdx.x & 31;
  const int s = src[e];
  const int d = dst[e];

  const float4* qs = (const float4*)(q + (size_t)s * INNER);
  const float4* kd = (const float4*)(k + (size_t)d * INNER);
  const float4 a0 = qs[2 * lane], a1 = qs[2 * lane + 1];
  const float4 b0 = kd[2 * lane], b1 = kd[2 * lane + 1];

  float p = a0.x * b0.x + a0.y * b0.y + a0.z * b0.z + a0.w * b0.w +
            a1.x * b1.x + a1.y * b1.y + a1.z * b1.z + a1.w * b1.w;
  p += __shfl_xor(p, 1);   // reduce 4 lanes of each head
  p += __shfl_xor(p, 2);

  if ((lane & 3) == 0) {
    const int h = lane >> 2;
    scores[(size_t)e * HEADS + h] = p;
    atomicMaxF32(mmax + (size_t)d * HEADS + h, p);
  }
}

// ---------------------------------------------------------------------------
// Edge exp: thread per (edge, head).  e = exp(s - m[dst]); z[dst] += e.
// ---------------------------------------------------------------------------
__global__ void __launch_bounds__(256)
gta_edge_exp_kernel(const int* __restrict__ dst, float* __restrict__ scores,
                    const float* __restrict__ mmax, float* __restrict__ zsum) {
  const int t = blockIdx.x * blockDim.x + threadIdx.x;
  if (t >= N_EDGES * HEADS) return;
  const int e = t >> 3;
  const int h = t & 7;
  const int d = dst[e];
  const float mv = mmax[(size_t)d * HEADS + h];   // finite: d has >=1 edge
  const float ex = __expf(scores[t] - mv);
  scores[t] = ex;
  atomicAdd(zsum + (size_t)d * HEADS + h, ex);
}

// ---------------------------------------------------------------------------
// Weighted scatter: wave per edge.  p = e / max(z[dst], 1e-16);
// out[dst][:] += v[src][:] * p[head] via global_atomic_add_f32.
// ---------------------------------------------------------------------------
__global__ void __launch_bounds__(256)
gta_edge_scatter_kernel(const float* __restrict__ vmat,
                        const int* __restrict__ src, const int* __restrict__ dst,
                        const float* __restrict__ scores,
                        const float* __restrict__ zsum,
                        float* __restrict__ out) {
  const int e = blockIdx.x * (blockDim.x >> 5) + (threadIdx.x >> 5);
  if (e >= N_EDGES) return;
  const int lane = threadIdx.x & 31;
  const int s = src[e];
  const int d = dst[e];
  const int h = lane >> 2;                 // dims 8*lane..8*lane+7 share head

  const float zz = zsum[(size_t)d * HEADS + h];
  const float p  = scores[(size_t)e * HEADS + h] / fmaxf(zz, 1e-16f);

  const float4* vp = (const float4*)(vmat + (size_t)s * INNER);
  const float4 v0 = vp[2 * lane], v1 = vp[2 * lane + 1];

  float* op = out + (size_t)d * INNER + lane * 8;
  atomicAdd(op + 0, v0.x * p);
  atomicAdd(op + 1, v0.y * p);
  atomicAdd(op + 2, v0.z * p);
  atomicAdd(op + 3, v0.w * p);
  atomicAdd(op + 4, v1.x * p);
  atomicAdd(op + 5, v1.y * p);
  atomicAdd(op + 6, v1.z * p);
  atomicAdd(op + 7, v1.w * p);
}

// ---------------------------------------------------------------------------
// Launch
// ---------------------------------------------------------------------------
extern "C" void kernel_launch(void* const* d_in, const int* in_sizes, int n_in,
                              void* d_out, int out_size, void* d_ws, size_t ws_size,
                              hipStream_t stream) {
  const float* x  = (const float*)d_in[0];
  const float* Wq = (const float*)d_in[1];
  const float* bq = (const float*)d_in[2];
  const float* Wk = (const float*)d_in[3];
  const float* bk = (const float*)d_in[4];
  const float* Wv = (const float*)d_in[5];
  const float* bv = (const float*)d_in[6];
  const int*  src = (const int*)d_in[7];
  const int*  dst = (const int*)d_in[8];
  float*      out = (float*)d_out;

  // Workspace layout (all offsets keep >=256B alignment):
  //   f32:  q | k | v | scores | m | z | bsc
  //   bf16: xb | Wt
  float* q      = (float*)d_ws;
  float* k      = q + (size_t)N_NODES * INNER;          // 12.8M f32
  float* v      = k + (size_t)N_NODES * INNER;          // 12.8M f32
  float* scores = v + (size_t)N_NODES * INNER;          // 12.8M f32
  float* mmax   = scores + (size_t)N_EDGES * HEADS;     // 6.4M f32
  float* zsum   = mmax + (size_t)N_NODES * HEADS;       // 0.4M f32
  float* bsc    = zsum + (size_t)N_NODES * HEADS;       // 0.4M f32
  __bf16* xb    = (__bf16*)(bsc + 1024);                // 12.8M bf16
  __bf16* Wt    = xb + (size_t)N_NODES * DIM;           // 196608 bf16
  (void)ws_size; (void)in_sizes; (void)n_in; (void)out_size;

  // 0) init out / m / z
  {
    const int total = N_NODES * INNER;                  // 12.8M (>= N*H)
    gta_init_kernel<<<(total + 255) / 256, 256, 0, stream>>>(out, mmax, zsum);
  }

  // 1) bf16 repack of x and (transposed, scale-folded) weights
  {
    const int quads = (N_NODES * DIM) / 4;              // 3.2M
    gta_cvt_x_kernel<<<(quads + 255) / 256, 256, 0, stream>>>(x, xb);
    const int wtot = 3 * DIM * INNER;                   // 196608
    gta_prep_w_kernel<<<(wtot + 255) / 256, 256, 0, stream>>>(Wq, bq, Wk, bk,
                                                              Wv, bv, Wt, bsc);
  }

  // 2) fused QKV projections (WMMA bf16; 4x b128 + 1 wmma per K-step)
  {
    dim3 grid(N_NODES / 16, (INNER / 16) / 4, 3);       // (3125, 4, 3)
    gta_qkv_gemm_kernel<<<grid, 128, 0, stream>>>(xb, Wt, bsc, q, k, v);
  }

  // 3) edge scores + segment max (wave per edge)
  {
    const int wavesPerBlock = 8;
    gta_edge_scores_kernel<<<(N_EDGES + wavesPerBlock - 1) / wavesPerBlock,
                             32 * wavesPerBlock, 0, stream>>>(q, k, src, dst,
                                                              scores, mmax);
  }

  // 4) exp + segment sum (thread per edge-head)
  {
    const int total = N_EDGES * HEADS;                  // 6.4M
    gta_edge_exp_kernel<<<(total + 255) / 256, 256, 0, stream>>>(dst, scores,
                                                                 mmax, zsum);
  }

  // 5) normalize + weighted scatter (wave per edge)
  {
    const int wavesPerBlock = 8;
    gta_edge_scatter_kernel<<<(N_EDGES + wavesPerBlock - 1) / wavesPerBlock,
                              32 * wavesPerBlock, 0, stream>>>(v, src, dst,
                                                               scores, zsum, out);
  }
}
